// LiteCNNMoEClassifier_42700564857411
// MI455X (gfx1250) — matrix operand order repaired and checked by reading, not compiled, not run on hardware
//
#include <hip/hip_runtime.h>
#include <hip/hip_bf16.h>

typedef __attribute__((ext_vector_type(16))) _Float16 v16h;
typedef __attribute__((ext_vector_type(8)))  float    v8f;
typedef __attribute__((ext_vector_type(8)))  unsigned v8u;

#define WMMA_F16(a, b, c) \
  __builtin_amdgcn_wmma_f32_16x16x32_f16(false, (a), false, (b), (short)0, (c), false, false)

// ---------------------------------------------------------------------------
// Fragment loaders (CDNA5 wave32 WMMA 16x16x32 f16 layouts per ISA 7.12.2)
// A 16x32 (MxK), row-major source, lane holds row m=lane&15:
//   VGPR v holds K = (v<4?0:16) + (lane>=16?8:0) + 2*(v&3) .. +1
__device__ __forceinline__ v16h frag_ldA(const _Float16* __restrict__ tile, int ldk) {
  const int lane = threadIdx.x & 31;
  const unsigned* row = (const unsigned*)(tile + (size_t)(lane & 15) * ldk);
  const int kh8 = (lane >> 4) << 3;
  v8u t;
#pragma unroll
  for (int v = 0; v < 8; ++v) {
    int k = ((v & 4) << 2) + kh8 + ((v & 3) << 1);
    t[v] = row[k >> 1];
  }
  return __builtin_bit_cast(v16h, t);
}

// B 32x16 (KxN) from an [N][K] row-major weight buffer; lane holds column
// n=lane&15, VGPR v holds K = (v<4?0:16) + (lane>=16?8:0) + 2*(v&3) .. +1
// (same K order as A).
__device__ __forceinline__ v16h frag_ldB(const _Float16* __restrict__ wnk, int ldk) {
  const int lane = threadIdx.x & 31;
  const unsigned* row =
      (const unsigned*)(wnk + (size_t)(lane & 15) * ldk);
  const int kh8 = (lane >> 4) << 3;
  v8u t;
#pragma unroll
  for (int v = 0; v < 8; ++v) t[v] = row[(((v & 4) << 2) + kh8 + ((v & 3) << 1)) >> 1];
  return __builtin_bit_cast(v16h, t);
}

// ---------------------------------------------------------------------------
// Weight prep kernels (f32 -> f16, layout for WMMA B operand)
__global__ void pack_conv_w_k(const float* __restrict__ w, _Float16* __restrict__ out,
                              int N, int Cin, int Kpad) {
  int i = blockIdx.x * blockDim.x + threadIdx.x;
  if (i >= N * Kpad) return;
  int n = i / Kpad, k = i - n * Kpad;
  int K = Cin * 9;
  out[i] = (k < K) ? (_Float16)w[(size_t)n * K + k] : (_Float16)0.f;
}

__global__ void transpose_pack_k(const float* __restrict__ w, _Float16* __restrict__ out,
                                 int E, int R, int C) {  // [E][R][C] -> [E][C][R] f16
  size_t i = (size_t)blockIdx.x * blockDim.x + threadIdx.x;
  size_t total = (size_t)E * R * C;
  if (i >= total) return;
  size_t rc = i % ((size_t)R * C);
  int e = (int)(i / ((size_t)R * C));
  int r = (int)(rc / C), c = (int)(rc % C);
  out[(size_t)e * R * C + (size_t)c * R + r] = (_Float16)w[i];
}

__global__ void pack_head_k(const float* __restrict__ w, _Float16* __restrict__ out,
                            int N, int Npad, int K) {
  int i = blockIdx.x * blockDim.x + threadIdx.x;
  if (i >= Npad * K) return;
  int n = i / K, k = i - n * K;
  out[i] = (n < N) ? (_Float16)w[(size_t)n * K + k] : (_Float16)0.f;
}

// ---------------------------------------------------------------------------
// Implicit-GEMM conv + fused BN/ReLU.  One wave: 16 output pixels x (16*NT) channels.
template <typename TIN, int NT>
__global__ __launch_bounds__(256) void conv_wmma_k(
    const TIN* __restrict__ in, const _Float16* __restrict__ wp,
    const float* __restrict__ cb, const float* __restrict__ bng,
    const float* __restrict__ bnb, _Float16* __restrict__ out,
    int Cin, int Hin, int Win, int Ho, int Wo, int Cout, int stride, int Kpad, int nwn) {
  const int wid = (blockIdx.x * blockDim.x + threadIdx.x) >> 5;
  const int lane = threadIdx.x & 31;
  const int mtile = wid / nwn;
  const int nbase = (wid - mtile * nwn) * (16 * NT);
  const int HW = Ho * Wo;
  // A-row coordinates for this lane (row m = lane&15 of the tile)
  const int ma = mtile * 16 + (lane & 15);
  const int ba = ma / HW;
  const int pa = ma - ba * HW;
  const int oha = pa / Wo, owa = pa - (pa / Wo) * Wo;
  const int kh8 = (lane >> 4) << 3;

  v8f acc[NT] = {};

  for (int kb = 0; kb < Kpad; kb += 32) {
    v16h a;
#pragma unroll
    for (int v = 0; v < 8; ++v) {
      const int kbase = kb + ((v & 4) << 2) + kh8 + ((v & 3) << 1);
#pragma unroll
      for (int j = 0; j < 2; ++j) {
        const int k = kbase + j;
        const int c = k / 9;
        const int rem = k - 9 * c;
        const int r = rem / 3;
        const int s = rem - 3 * r;
        const int ih = oha * stride + r - 1;
        const int iw = owa * stride + s - 1;
        _Float16 val = (_Float16)0.f;
        if (c < Cin && (unsigned)ih < (unsigned)Hin && (unsigned)iw < (unsigned)Win)
          val = (_Float16)(float)in[(((size_t)ba * Cin + c) * Hin + ih) * Win + iw];
        a[2 * v + j] = val;
      }
    }
#pragma unroll
    for (int t = 0; t < NT; ++t) {
      v16h bf = frag_ldB(wp + (size_t)(nbase + t * 16) * Kpad + kb, Kpad);
      acc[t] = WMMA_F16(a, bf, acc[t]);
    }
  }

  const int ncol = lane & 15;
  const int mof = (lane >> 4) << 3;
  const float inv = rsqrtf(1.0f + 1e-5f);
#pragma unroll
  for (int t = 0; t < NT; ++t) {
    const int co = nbase + t * 16 + ncol;
    const float scale = bng[co] * inv;
    const float bias = cb[co];
    const float shf = bnb[co];
#pragma unroll
    for (int v = 0; v < 8; ++v) {
      const int m = mtile * 16 + v + mof;
      const int b = m / HW;
      const int p = m - b * HW;
      float y = (acc[t][v] + bias) * scale + shf;
      y = y > 0.f ? y : 0.f;
      out[((size_t)b * Cout + co) * HW + p] = (_Float16)y;
    }
  }
}

// ---------------------------------------------------------------------------
// Global avg-pool over 8x8 + f32/f16 feature write.
__global__ void avgpool_k(const _Float16* __restrict__ h3, float* __restrict__ feats,
                          _Float16* __restrict__ featsh, int total, int P) {
  int i = blockIdx.x * blockDim.x + threadIdx.x;
  if (i >= total) return;
  const _Float16* p = h3 + (size_t)i * P;
  float s = 0.f;
  for (int j = 0; j < P; ++j) s += (float)p[j];
  s *= (1.f / (float)P);
  feats[i] = s;
  featsh[i] = (_Float16)s;
}

// ---------------------------------------------------------------------------
// Router: logits -> top-2 -> softmax -> scattered gates [B][16]. One thread/sample.
__global__ void router_k(const float* __restrict__ feats, const float* __restrict__ rw,
                         const float* __restrict__ rb, float* __restrict__ gates,
                         int M, int D) {
  int b = blockIdx.x * blockDim.x + threadIdx.x;
  if (b >= M) return;
  float logits[16];
  for (int e = 0; e < 16; ++e) {
    float s = rb[e];
    const float* fe = feats + (size_t)b * D;
    const float* we = rw + (size_t)e * D;
    for (int d = 0; d < D; ++d) s += fe[d] * we[d];
    logits[e] = s;
  }
  int i0 = 0;
  for (int e = 1; e < 16; ++e) if (logits[e] > logits[i0]) i0 = e;
  int i1 = (i0 == 0) ? 1 : 0;
  for (int e = 0; e < 16; ++e)
    if (e != i0 && logits[e] > logits[i1]) i1 = e;
  float l0 = logits[i0], l1 = logits[i1];
  float e0 = __expf(0.f), e1 = __expf(l1 - l0);  // e0 = 1
  float inv = 1.f / (e0 + e1);
  for (int e = 0; e < 16; ++e) gates[(size_t)b * 16 + e] = 0.f;
  gates[(size_t)b * 16 + i0] = e0 * inv;
  gates[(size_t)b * 16 + i1] = e1 * inv;
}

// ---------------------------------------------------------------------------
// Expert layer 1: per-expert GEMM with fused bias+ReLU -> f16. blockIdx.y = expert.
template <int NT>
__global__ __launch_bounds__(256) void gemm_bias_relu_k(
    const _Float16* __restrict__ A, const _Float16* __restrict__ Bw,
    const float* __restrict__ bias, _Float16* __restrict__ C,
    int N, int K, size_t bStrideE, size_t biasStrideE, size_t cStrideE) {
  const int e = blockIdx.y;
  Bw += (size_t)e * bStrideE;
  bias += (size_t)e * biasStrideE;
  C += (size_t)e * cStrideE;
  const int wid = (blockIdx.x * blockDim.x + threadIdx.x) >> 5;
  const int lane = threadIdx.x & 31;
  const int wn = N / (16 * NT);
  const int mtile = wid / wn;
  const int nbase = (wid - mtile * wn) * (16 * NT);
  v8f acc[NT] = {};
  const _Float16* Arow = A + (size_t)mtile * 16 * K;
  for (int kb = 0; kb < K; kb += 32) {
    v16h a = frag_ldA(Arow + kb, K);
#pragma unroll
    for (int t = 0; t < NT; ++t) {
      v16h b = frag_ldB(Bw + (size_t)(nbase + t * 16) * K + kb, K);
      acc[t] = WMMA_F16(a, b, acc[t]);
    }
  }
  const int ncol = lane & 15, mof = (lane >> 4) << 3;
#pragma unroll
  for (int t = 0; t < NT; ++t) {
    const int n = nbase + t * 16 + ncol;
    const float bv = bias[n];
#pragma unroll
    for (int v = 0; v < 8; ++v) {
      const int m = mtile * 16 + v + mof;
      float y = acc[t][v] + bv;
      y = y > 0.f ? y : 0.f;
      C[(size_t)m * N + n] = (_Float16)y;
    }
  }
}

// ---------------------------------------------------------------------------
// Expert layer 2 + gate-weighted mix across all 16 experts.
// Wave computes a 16(samples) x 16(D dims) tile of `mixed`.
__global__ __launch_bounds__(256) void moe_mix_k(
    const _Float16* __restrict__ hexp,   // [E][M][HD]
    const _Float16* __restrict__ w2t,    // [E][D][HD]   (NxK row-major per expert)
    const float* __restrict__ eb2,       // [E][D]
    const float* __restrict__ gates,     // [M][E]
    _Float16* __restrict__ mixed,        // [M][D]
    int M, int HD, int D) {
  const int wid = (blockIdx.x * blockDim.x + threadIdx.x) >> 5;
  const int lane = threadIdx.x & 31;
  const int ntiles = D / 16;
  const int mtile = wid / ntiles;
  const int nb = (wid - mtile * ntiles) * 16;
  const int col = nb + (lane & 15);
  const int mof = (lane >> 4) << 3;

  float facc[8] = {0.f, 0.f, 0.f, 0.f, 0.f, 0.f, 0.f, 0.f};
  for (int e = 0; e < 16; ++e) {
    v8f acc = {};
    const _Float16* Ae = hexp + ((size_t)e * M + (size_t)mtile * 16) * HD;
    const _Float16* Be = w2t + ((size_t)e * D + nb) * HD;
    for (int kb = 0; kb < HD; kb += 32) {
      v16h a = frag_ldA(Ae + kb, HD);
      v16h b = frag_ldB(Be + kb, HD);
      acc = WMMA_F16(a, b, acc);
    }
    const float b2 = eb2[(size_t)e * D + col];
#pragma unroll
    for (int v = 0; v < 8; ++v) {
      const int m = mtile * 16 + v + mof;
      facc[v] += gates[(size_t)m * 16 + e] * (acc[v] + b2);
    }
  }
#pragma unroll
  for (int v = 0; v < 8; ++v) {
    const int m = mtile * 16 + v + mof;
    mixed[(size_t)m * D + col] = (_Float16)facc[v];
  }
}

// ---------------------------------------------------------------------------
// Classifier head: f16 GEMM, f32 out, N guarded (1000 of 1008).
__global__ __launch_bounds__(256) void head_k(
    const _Float16* __restrict__ A, const _Float16* __restrict__ Wp,
    const float* __restrict__ hb, float* __restrict__ out,
    int N, int Npad, int K) {
  const int wid = (blockIdx.x * blockDim.x + threadIdx.x) >> 5;
  const int lane = threadIdx.x & 31;
  const int ntiles = Npad / 16;
  const int mtile = wid / ntiles;
  const int nb = (wid - mtile * ntiles) * 16;
  v8f acc = {};
  const _Float16* Arow = A + (size_t)mtile * 16 * K;
  const _Float16* Brow = Wp + (size_t)nb * K;
  for (int kb = 0; kb < K; kb += 32) {
    v16h a = frag_ldA(Arow + kb, K);
    v16h b = frag_ldB(Brow + kb, K);
    acc = WMMA_F16(a, b, acc);
  }
  const int col = nb + (lane & 15);
  const int mof = (lane >> 4) << 3;
  if (col < N) {
    const float bv = hb[col];
#pragma unroll
    for (int v = 0; v < 8; ++v) {
      const int m = mtile * 16 + v + mof;
      out[(size_t)m * N + col] = acc[v] + bv;
    }
  }
}

// ---------------------------------------------------------------------------
extern "C" void kernel_launch(void* const* d_in, const int* in_sizes, int n_in,
                              void* d_out, int out_size, void* d_ws, size_t ws_size,
                              hipStream_t stream) {
  const int B = 2048, D = 256, E = 16, HD = 1024, NC = 1000, NCP = 1008;
  const int K1 = 32 /*27 padded*/, K2 = 576, K3 = 1152;

  const float* x       = (const float*)d_in[0];
  const float* c1w     = (const float*)d_in[1];
  const float* c1b     = (const float*)d_in[2];
  const float* bn1g    = (const float*)d_in[3];
  const float* bn1b    = (const float*)d_in[4];
  const float* c2w     = (const float*)d_in[5];
  const float* c2b     = (const float*)d_in[6];
  const float* bn2g    = (const float*)d_in[7];
  const float* bn2b    = (const float*)d_in[8];
  const float* c3w     = (const float*)d_in[9];
  const float* c3b     = (const float*)d_in[10];
  const float* bn3g    = (const float*)d_in[11];
  const float* bn3b    = (const float*)d_in[12];
  const float* rw      = (const float*)d_in[13];
  const float* rb      = (const float*)d_in[14];
  const float* ew1     = (const float*)d_in[15];
  const float* eb1     = (const float*)d_in[16];
  const float* ew2     = (const float*)d_in[17];
  const float* eb2     = (const float*)d_in[18];
  const float* hw      = (const float*)d_in[19];
  const float* hb      = (const float*)d_in[20];
  float* out = (float*)d_out;

  // ---- workspace layout ----
  char* ws = (char*)d_ws;
  size_t off = 0;
  auto alloc = [&](size_t bytes) {
    void* p = ws + off;
    off = (off + bytes + 255) & ~(size_t)255;
    return p;
  };
  _Float16* w1p   = (_Float16*)alloc((size_t)64 * K1 * 2);
  _Float16* w2p   = (_Float16*)alloc((size_t)128 * K2 * 2);
  _Float16* w3p   = (_Float16*)alloc((size_t)256 * K3 * 2);
  _Float16* ew1t  = (_Float16*)alloc((size_t)E * HD * D * 2);   // [E][HD][D]
  _Float16* ew2t  = (_Float16*)alloc((size_t)E * D * HD * 2);   // [E][D][HD]
  _Float16* headp = (_Float16*)alloc((size_t)NCP * D * 2);
  _Float16* h1    = (_Float16*)alloc((size_t)B * 64 * 32 * 32 * 2);
  _Float16* h2    = (_Float16*)alloc((size_t)B * 128 * 16 * 16 * 2);
  _Float16* h3    = (_Float16*)alloc((size_t)B * 256 * 8 * 8 * 2);
  float*    feats = (float*)alloc((size_t)B * D * 4);
  _Float16* featsh= (_Float16*)alloc((size_t)B * D * 2);
  float*    gates = (float*)alloc((size_t)B * E * 4);
  _Float16* hexp  = (_Float16*)alloc((size_t)E * B * HD * 2);   // [E][B][HD]
  _Float16* mixedh= (_Float16*)alloc((size_t)B * D * 2);
  (void)ws_size; (void)n_in; (void)in_sizes; (void)out_size;

  const int T = 256;
  auto cdiv = [](long long a, long long b) { return (int)((a + b - 1) / b); };

  // ---- weight prep ----
  pack_conv_w_k<<<cdiv(64 * K1, T), T, 0, stream>>>(c1w, w1p, 64, 3, K1);
  pack_conv_w_k<<<cdiv(128 * K2, T), T, 0, stream>>>(c2w, w2p, 128, 64, K2);
  pack_conv_w_k<<<cdiv(256 * K3, T), T, 0, stream>>>(c3w, w3p, 256, 128, K3);
  transpose_pack_k<<<cdiv((long long)E * D * HD, T), T, 0, stream>>>(ew1, ew1t, E, D, HD);
  transpose_pack_k<<<cdiv((long long)E * HD * D, T), T, 0, stream>>>(ew2, ew2t, E, HD, D);
  pack_head_k<<<cdiv(NCP * D, T), T, 0, stream>>>(hw, headp, NC, NCP, D);

  // ---- conv stem (implicit GEMM, WMMA f16) ----
  // conv1: M = B*32*32 -> 131072 m-tiles, NT=4 covers all 64 channels
  conv_wmma_k<float, 4><<<(B * 1024 / 16) / 8, T, 0, stream>>>(
      x, w1p, c1b, bn1g, bn1b, h1, 3, 32, 32, 32, 32, 64, 1, K1, 1);
  // conv2: M = B*16*16, NT=8 covers 128 channels
  conv_wmma_k<_Float16, 8><<<(B * 256 / 16) / 8, T, 0, stream>>>(
      h1, w2p, c2b, bn2g, bn2b, h2, 64, 32, 32, 16, 16, 128, 2, K2, 1);
  // conv3: M = B*8*8, NT=8, 2 wave-groups cover 256 channels
  conv_wmma_k<_Float16, 8><<<(B * 64 / 16) * 2 / 8, T, 0, stream>>>(
      h2, w3p, c3b, bn3g, bn3b, h3, 128, 16, 16, 8, 8, 256, 2, K3, 2);

  // ---- pool + router ----
  avgpool_k<<<cdiv((long long)B * D, T), T, 0, stream>>>(h3, feats, featsh, B * D, 64);
  router_k<<<cdiv(B, T), T, 0, stream>>>(feats, rw, rb, gates, B, D);

  // ---- experts ----
  // E1: per-expert [B,256] x [256,1024] -> hexp, 2048 waves per expert
  {
    dim3 grid((B / 16) * (HD / 64) / 8, E);
    gemm_bias_relu_k<4><<<grid, T, 0, stream>>>(
        featsh, ew1t, eb1, hexp, HD, D,
        (size_t)HD * D, (size_t)HD, (size_t)B * HD);
  }
  // E2 + gating + mix: (B/16)*(D/16) = 2048 waves
  moe_mix_k<<<(B / 16) * (D / 16) / 8, T, 0, stream>>>(
      hexp, ew2t, eb2, gates, mixedh, B, HD, D);

  // ---- head ----
  head_k<<<(B / 16) * (NCP / 16) / 8, T, 0, stream>>>(
      mixedh, headp, hb, out, NC, NCP, D);
}
